// ACNN_50345606643882
// MI455X (gfx1250) — matrix operand order repaired and checked by reading, not compiled, not run on hardware
//
#include <hip/hip_runtime.h>
#include <hip/hip_bf16.h>
#include <math.h>

typedef __attribute__((ext_vector_type(16))) _Float16 v16h;
typedef __attribute__((ext_vector_type(8)))  float    v8f;

#define NB    128
#define NTOK  100
#define DWE   300
#define DPE   50
#define DDIM  400
#define DPAD  416      // 400 padded to multiple of 32 (K-chunks)
#define ROWS  116      // row r = token n+1; row 0 and rows 101..115 are zero pad
#define DCN   64
#define KW    3

// ---- dynamic LDS layout (byte offsets) ----
#define XC_BYTES   (ROWS*DPAD*2)     // 96512  : x_concat f16, padded
#define FSEC_F     964               // e1(300) e2(300) A1(100) A2(100) alpha(100) resi(64)
#define FSEC_BYTES (FSEC_F*4)        // 3856
#define RT_BYTES   (112*64*2)        // 14336  : R_star^T [n][c] f16
#define RC_BYTES   (64*128*2)        // 16384  : R_star   [c][n] f16, n padded to 128
#define G_BYTES    (112*64*4)        // 28672  : G f32 (also resiPart / wo_full scratch)
#define AP_BYTES   (128*64*2)        // 16384  : AP f16 [n][c'], n padded to 128
#define SMEM_BYTES (XC_BYTES+FSEC_BYTES+RT_BYTES+RC_BYTES+G_BYTES+AP_BYTES) // 176144

// ---- WMMA fragment loaders (CDNA5 16x16x32 f16 layouts, wave32) ----
// A 16x32 f16: lane l holds row M=l&15; VGPR0-3 -> K = (l<16?0:8)+0..7, VGPR4-7 -> +16.
__device__ inline v16h loadA16(const _Float16* base, int ld, int m0, int kb, int lane) {
    const _Float16* p = base + (size_t)(m0 + (lane & 15)) * ld + kb + ((lane >> 4) << 3);
    union { v16h v; uint4 q[2]; } u;
    u.q[0] = *(const uint4*)(p);        // K group 0..7
    u.q[1] = *(const uint4*)(p + 16);   // K group 16..23
    return u.v;
}
// B 32x16 f16, stored K-major (row=K, N contiguous): lane l holds N=n0+(l&15), K = (l<16?0:8)+0..7, then +16.
__device__ inline v16h loadB16(const _Float16* base, int ld, int n0, int kb, int lane) {
    const _Float16* p = base + (size_t)(kb + ((lane >> 4) << 3)) * ld + n0 + (lane & 15);
    v16h v;
    #pragma unroll
    for (int i = 0; i < 8; ++i) {
        v[i]     = p[(size_t)i * ld];
        v[i + 8] = p[(size_t)(16 + i) * ld];
    }
    return v;
}
__device__ inline v8f wmma16(v16h a, v16h b, v8f c) {
    return __builtin_amdgcn_wmma_f32_16x16x32_f16(false, a, false, b, (short)0, c, false, false);
}
__device__ inline float wred_sum(float v) {
    for (int o = 16; o; o >>= 1) v += __shfl_xor(v, o, 32);
    return v;
}
__device__ inline float wred_max(float v) {
    for (int o = 16; o; o >>= 1) v = fmaxf(v, __shfl_xor(v, o, 32));
    return v;
}

// ---- weight-prep kernels (run once per launch, shared across all 128 blocks) ----
__global__ void prep_w2(const float* __restrict__ U, const float* __restrict__ y_emb,
                        _Float16* __restrict__ W2h) {
    int i = blockIdx.x * blockDim.x + threadIdx.x;       // 64*64
    if (i >= 64 * 64) return;
    int c = i >> 6, cp = i & 63;
    float s = 0.f;
    for (int r = 0; r < 64; ++r) s += U[c * 64 + r] * y_emb[r * 64 + cp];
    W2h[i] = (_Float16)s;                                 // K-major: [c][c']
}
__global__ void prep_bres(const float* __restrict__ residual, _Float16* __restrict__ Bres) {
    int i = blockIdx.x * blockDim.x + threadIdx.x;       // 416*64
    if (i >= DPAD * DCN) return;
    int k = i >> 6, c = i & 63;
    Bres[i] = (_Float16)(k < DDIM ? residual[k * 64 + c] : 0.f);   // K-major: [d][c]
}
__global__ void prep_bconv(const float* __restrict__ conv_w, _Float16* __restrict__ Bconv) {
    int i = blockIdx.x * blockDim.x + threadIdx.x;       // 3*416*64
    if (i >= KW * DPAD * DCN) return;
    int kk = i >> 6, c = i & 63;
    int kw = kk / DPAD, d = kk - kw * DPAD;
    Bconv[i] = (_Float16)(d < DDIM ? conv_w[(c * KW + kw) * DDIM + d] : 0.f); // [kw*416+d][c]
}
__global__ void copy_yemb(const float* __restrict__ y, float* __restrict__ out) {
    int i = blockIdx.x * blockDim.x + threadIdx.x;
    if (i < 64 * 64) out[NB * 64 + i] = y[i];            // second tuple output
}

// ---- fully fused per-batch kernel: 1 block = 1 batch element, LDS-resident ----
__global__ __launch_bounds__(256) void acnn_fused(
    const int* __restrict__ x, const int* __restrict__ e1, const int* __restrict__ e2,
    const int* __restrict__ dist1, const int* __restrict__ dist2,
    const float* __restrict__ other_emb, const float* __restrict__ dist_emb,
    const float* __restrict__ conv_b,
    const _Float16* __restrict__ W2h, const _Float16* __restrict__ Bres,
    const _Float16* __restrict__ Bconv, float* __restrict__ out)
{
    extern __shared__ char smem[];
    _Float16* xc   = (_Float16*)smem;
    float*    fsec = (float*)(smem + XC_BYTES);
    float *e1s = fsec, *e2s = fsec + 300, *A1 = fsec + 600, *A2 = fsec + 700,
          *alpha = fsec + 800, *resiV = fsec + 900;
    _Float16* RT = (_Float16*)(smem + XC_BYTES + FSEC_BYTES);
    _Float16* RC = (_Float16*)((char*)RT + RT_BYTES);
    float*    Gm = (float*)((char*)RC + RC_BYTES);
    _Float16* AP = (_Float16*)((char*)Gm + G_BYTES);

    const int b = blockIdx.x;
    const int tid = threadIdx.x;
    const int wave = tid >> 5, lane = tid & 31;

    // Phase 1: zero pad rows, gather embeddings (f16), fused attention dots (f32)
    for (int i = tid; i < 16 * DPAD; i += 256) {
        int rr = i / DPAD, d = i - rr * DPAD;
        int row = (rr == 0) ? 0 : (100 + rr);
        xc[row * DPAD + d] = (_Float16)0.f;
    }
    {
        int i1 = e1[b], i2 = e2[b];
        for (int d = tid; d < DWE; d += 256) {
            e1s[d] = (i1 == 0) ? 0.f : other_emb[(size_t)(i1 - 1) * DWE + d];
            e2s[d] = (i2 == 0) ? 0.f : other_emb[(size_t)(i2 - 1) * DWE + d];
        }
    }
    __syncthreads();
    for (int n = wave; n < NTOK; n += 8) {
        int xi = x[b * NTOK + n];
        const float* xrow = (xi == 0) ? nullptr : other_emb + (size_t)(xi - 1) * DWE;
        _Float16* dst = xc + (n + 1) * DPAD;
        float s1 = 0.f, s2 = 0.f;
        for (int d = lane; d < DWE; d += 32) {
            float v = xrow ? xrow[d] : 0.f;
            s1 += v * e1s[d];
            s2 += v * e2s[d];
            dst[d] = (_Float16)v;
        }
        s1 = wred_sum(s1); s2 = wred_sum(s2);
        if (lane == 0) { A1[n] = s1; A2[n] = s2; }
        int d1i = dist1[b * NTOK + n], d2i = dist2[b * NTOK + n];
        for (int d = lane; d < DPE; d += 32) {
            dst[DWE + d]       = (_Float16)dist_emb[d1i * DPE + d];
            dst[DWE + DPE + d] = (_Float16)dist_emb[d2i * DPE + d];
        }
        if (lane < 16) dst[DDIM + lane] = (_Float16)0.f;   // K pad 400..415
    }
    __syncthreads();

    // Phase 2: alpha = 0.5*(softmax(A1)+softmax(A2)) over n
    if (wave == 0) {
        float m1 = -3.4e38f, m2 = -3.4e38f;
        for (int n = lane; n < NTOK; n += 32) { m1 = fmaxf(m1, A1[n]); m2 = fmaxf(m2, A2[n]); }
        m1 = wred_max(m1); m2 = wred_max(m2);
        float s1 = 0.f, s2 = 0.f;
        for (int n = lane; n < NTOK; n += 32) { s1 += expf(A1[n] - m1); s2 += expf(A2[n] - m2); }
        s1 = wred_sum(s1); s2 = wred_sum(s2);
        float r1 = 1.f / s1, r2 = 1.f / s2;
        for (int n = lane; n < NTOK; n += 32)
            alpha[n] = 0.5f * (expf(A1[n] - m1) * r1 + expf(A2[n] - m2) * r2);
    }
    if (tid < 64) resiV[tid] = -3.4e38f;
    __syncthreads();

    // Phase 3: resi GEMM: x_concat(100x416) @ Bres(416x64), row-max -> resiV[64]
    for (int t = wave; t < 28; t += 8) {
        int mt = t >> 2, m0 = mt << 4, n0 = (t & 3) << 4;
        v8f acc = {};
        for (int kb = 0; kb < DPAD; kb += 32) {
            v16h a  = loadA16(xc + DPAD, DPAD, m0, kb, lane);   // token n -> row n+1
            v16h bb = loadB16(Bres, DCN, n0, kb, lane);
            acc = wmma16(a, bb, acc);
        }
        int hb = (lane >> 4) << 3, nn = lane & 15;
        float mx = -3.4e38f;
        #pragma unroll
        for (int r = 0; r < 8; ++r) {
            int n = m0 + hb + r;
            if (n < NTOK) mx = fmaxf(mx, acc[r]);
        }
        mx = fmaxf(mx, __shfl_xor(mx, 16, 32));  // combine lane halves (same N column)
        if (lane < 16) Gm[mt * 64 + n0 + nn] = mx;
    }
    __syncthreads();
    if (tid < 64) {
        float mx = -3.4e38f;
        #pragma unroll
        for (int mt = 0; mt < 7; ++mt) mx = fmaxf(mx, Gm[mt * 64 + tid]);
        resiV[tid] = mx;
    }
    // Phase 4: fold alpha into xc (in place, f16)
    for (int i = tid; i < NTOK * DPAD; i += 256) {
        int n = i / DPAD, d = i - n * DPAD;
        _Float16* p = xc + (n + 1) * DPAD + d;
        *p = (_Float16)((float)(*p) * alpha[n]);
    }
    __syncthreads();

    // Phase 5: conv-as-GEMM (K=3x416) + bias + tanh -> R_star (both layouts, f16)
    for (int t = wave; t < 28; t += 8) {
        int m0 = (t >> 2) << 4, n0 = (t & 3) << 4;
        v8f acc = {};
        #pragma unroll
        for (int kw = 0; kw < KW; ++kw) {
            const _Float16* Bk = Bconv + (size_t)kw * DPAD * DCN;
            for (int kb = 0; kb < DPAD; kb += 32) {
                v16h a  = loadA16(xc + kw * DPAD, DPAD, m0, kb, lane); // window row n+kw-1
                v16h bb = loadB16(Bk, DCN, n0, kb, lane);
                acc = wmma16(a, bb, acc);
            }
        }
        int hb = (lane >> 4) << 3, nn = lane & 15;
        int c = n0 + nn;
        float bias = conv_b[c];
        #pragma unroll
        for (int r = 0; r < 8; ++r) {
            int n = m0 + hb + r;
            float v = (n < NTOK) ? tanhf(acc[r] + bias) : 0.f;
            RT[n * 64 + c]  = (_Float16)v;
            RC[c * 128 + n] = (_Float16)v;
        }
    }
    for (int i = tid; i < 64 * 16; i += 256)     // zero RC n-pad 112..127
        RC[(i >> 4) * 128 + 112 + (i & 15)] = (_Float16)0.f;
    __syncthreads();

    // Phase 6: G = R_star^T(100x64) @ W2(64x64)
    for (int t = wave; t < 28; t += 8) {
        int m0 = (t >> 2) << 4, n0 = (t & 3) << 4;
        v8f acc = {};
        for (int kb = 0; kb < 64; kb += 32) {
            v16h a  = loadA16(RT, 64, m0, kb, lane);
            v16h bb = loadB16(W2h, DCN, n0, kb, lane);
            acc = wmma16(a, bb, acc);
        }
        int hb = (lane >> 4) << 3, nn = lane & 15;
        #pragma unroll
        for (int r = 0; r < 8; ++r) {
            int n = m0 + hb + r;
            if (n < NTOK) Gm[n * 64 + n0 + nn] = acc[r];
        }
    }
    __syncthreads();

    // Phase 7: column softmax over n -> AP (f16, rows 100..127 zero)
    for (int c = wave; c < 64; c += 8) {
        float mx = -3.4e38f;
        for (int n = lane; n < NTOK; n += 32) mx = fmaxf(mx, Gm[n * 64 + c]);
        mx = wred_max(mx);
        float s = 0.f;
        for (int n = lane; n < NTOK; n += 32) s += expf(Gm[n * 64 + c] - mx);
        s = wred_sum(s);
        float inv = 1.f / s;
        for (int n = lane; n < NTOK; n += 32)
            AP[n * 64 + c] = (_Float16)(expf(Gm[n * 64 + c] - mx) * inv);
    }
    for (int i = tid; i < 28 * 64; i += 256)
        AP[(NTOK + i / 64) * 64 + (i & 63)] = (_Float16)0.f;
    __syncthreads();

    // Phase 8: wo_full = R_star(64x128) @ AP(128x64) -> Gm; then max over c' + resi
    for (int t = wave; t < 16; t += 8) {
        int m0 = (t >> 2) << 4, n0 = (t & 3) << 4;
        v8f acc = {};
        for (int kb = 0; kb < 128; kb += 32) {
            v16h a  = loadA16(RC, 128, m0, kb, lane);
            v16h bb = loadB16(AP, 64, n0, kb, lane);
            acc = wmma16(a, bb, acc);
        }
        int hb = (lane >> 4) << 3, nn = lane & 15;
        #pragma unroll
        for (int r = 0; r < 8; ++r)
            Gm[(m0 + hb + r) * 64 + n0 + nn] = acc[r];
    }
    __syncthreads();
    if (tid < 64) {
        float mx = -3.4e38f;
        #pragma unroll
        for (int k = 0; k < 64; ++k) mx = fmaxf(mx, Gm[tid * 64 + k]);
        out[b * 64 + tid] = mx + resiV[tid];
    }
}

extern "C" void kernel_launch(void* const* d_in, const int* in_sizes, int n_in,
                              void* d_out, int out_size, void* d_ws, size_t ws_size,
                              hipStream_t stream) {
    (void)in_sizes; (void)n_in; (void)out_size; (void)ws_size;
    const int*   x         = (const int*)d_in[0];
    const int*   e1        = (const int*)d_in[1];
    const int*   e2        = (const int*)d_in[2];
    const int*   dist1     = (const int*)d_in[3];
    const int*   dist2     = (const int*)d_in[4];
    const float* other_emb = (const float*)d_in[5];
    const float* dist_emb  = (const float*)d_in[6];
    const float* conv_w    = (const float*)d_in[7];
    const float* conv_b    = (const float*)d_in[8];
    const float* y_emb     = (const float*)d_in[9];
    const float* U         = (const float*)d_in[10];
    const float* residual  = (const float*)d_in[11];
    float* out = (float*)d_out;

    char* ws = (char*)d_ws;
    _Float16* W2h   = (_Float16*)ws;                      //  8192 B
    _Float16* BresW = (_Float16*)(ws + 8192);             // 53248 B
    _Float16* BconW = (_Float16*)(ws + 8192 + 53248);     // 159744 B

    prep_w2   <<<16, 256, 0, stream>>>(U, y_emb, W2h);
    prep_bres <<<(DPAD * DCN + 255) / 256, 256, 0, stream>>>(residual, BresW);
    prep_bconv<<<(KW * DPAD * DCN + 255) / 256, 256, 0, stream>>>(conv_w, BconW);
    copy_yemb <<<16, 256, 0, stream>>>(y_emb, out);
    acnn_fused<<<NB, 256, SMEM_BYTES, stream>>>(x, e1, e2, dist1, dist2, other_emb,
                                                dist_emb, conv_b, W2h, BresW, BconW, out);
}
